// LocalLearningUnit_49383533969489
// MI455X (gfx1250) — matrix operand order repaired and checked by reading, not compiled, not run on hardware
//
#include <hip/hip_runtime.h>

typedef _Float16 half_t;
typedef __attribute__((ext_vector_type(16))) _Float16 v16h;
typedef __attribute__((ext_vector_type(8)))  float    v8f;

#define N_RELS   8
#define CH       128                 // IN_CH == HID == 128
#define CT_TILES 8                   // 128 / 16 output-column tiles
#define KC_TILES 4                   // 128 / 32 K chunks
#define FRAGS_PER_MAT (CT_TILES * KC_TILES * 32)   // 1024 v16h (32 B each) per matrix

// ---------------------------------------------------------------------------
// Repack W[8][128][128] and root[128][128] (f32, row-major K x H) into the
// WMMA B-fragment order used here (f16, contiguous K per half-wave). One thread per (mat, ct, kc, lane):
//   lanes 0-15  hold K = k0 .. k0+15   (col = ct*16 + lane)
//   lanes 16-31 hold K = k0+16 .. k0+31
// so the GEMM's B load is a single contiguous 32-byte v16h load per lane.
// ---------------------------------------------------------------------------
__global__ void build_bfrag_kernel(const float* __restrict__ W,
                                   const float* __restrict__ root,
                                   v16h* __restrict__ frag) {
  int id = blockIdx.x * blockDim.x + threadIdx.x;
  const int total = (N_RELS + 1) * FRAGS_PER_MAT;
  if (id >= total) return;
  int lane = id & 31;
  int kc   = (id >> 5) & 3;
  int ct   = (id >> 7) & 7;
  int r    = id >> 10;                        // 0..7 = W[r], 8 = root
  const float* M = (r < N_RELS) ? (W + (size_t)r * CH * CH) : root;
  int col = ct * 16 + (lane & 15);
  int k0  = kc * 32 + ((lane & 16) ? 16 : 0);
  v16h v;
#pragma unroll
  for (int j = 0; j < 16; ++j)
    v[j] = (half_t)M[(size_t)(k0 + j) * CH + col];
  frag[id] = v;
}

// ---------------------------------------------------------------------------
// out[16-row tile, :] (+)= scale(A) @ B   using v_wmma_f32_16x16x32_f16.
//   A     : [N,128] f32 row-major (x for root term, agg for relation terms)
//   cnt   : per-row mean divisor (nullptr => scale 1, root term)
//   bias  : non-null => write mode (out = A@B + bias), null => out += A@B
//   doRelu: apply ReLU in the store path (final accumulation pass)
// One wave per 16-row tile, 8 waves per block. All 8 waves share the same
// B matrix, so the 32 KB of B fragments are staged once per block into LDS
// (ds_load_b128 instead of 8x redundant global traffic + load-wait stalls).
// ---------------------------------------------------------------------------
__global__ void __launch_bounds__(256)
rgcn_wmma_gemm(const float* __restrict__ A,
               const float* __restrict__ cnt,
               const v16h*  __restrict__ bfrag,   // FRAGS_PER_MAT entries
               const float* __restrict__ bias,
               float* __restrict__ out,
               int nRowTiles, int doRelu) {
  __shared__ v16h sfrag[FRAGS_PER_MAT];          // 32 KB
#pragma unroll
  for (int i = 0; i < FRAGS_PER_MAT / 256; ++i)
    sfrag[threadIdx.x + 256 * i] = bfrag[threadIdx.x + 256 * i];
  __syncthreads();

  const int lane = threadIdx.x & 31;
  const int wave = threadIdx.x >> 5;
  const int tile = blockIdx.x * 8 + wave;
  if (tile >= nRowTiles) return;               // wave-uniform: EXEC stays all-1s

  const int m0  = tile * 16;
  const int row = m0 + (lane & 15);            // A: both half-waves cover rows m0..m0+15
  float sc = 1.0f;
  if (cnt) sc = 1.0f / fmaxf(cnt[row], 1.0f);  // fold mean division into f16 conversion

  // A fragments: lane<16 holds K = k0..k0+15,
  //              lane>=16 holds K = k0+16..k0+31
  const int kshift = (lane & 16) ? 16 : 0;
  v16h afrag[KC_TILES];
#pragma unroll
  for (int kc = 0; kc < KC_TILES; ++kc) {
    const float4* p4 =
        reinterpret_cast<const float4*>(A + (size_t)row * CH + kc * 32 + kshift);
    float4 q0 = p4[0], q1 = p4[1], q2 = p4[2], q3 = p4[3];
    v16h a;
    a[0]  = (half_t)(q0.x * sc); a[1]  = (half_t)(q0.y * sc);
    a[2]  = (half_t)(q0.z * sc); a[3]  = (half_t)(q0.w * sc);
    a[4]  = (half_t)(q1.x * sc); a[5]  = (half_t)(q1.y * sc);
    a[6]  = (half_t)(q1.z * sc); a[7]  = (half_t)(q1.w * sc);
    a[8]  = (half_t)(q2.x * sc); a[9]  = (half_t)(q2.y * sc);
    a[10] = (half_t)(q2.z * sc); a[11] = (half_t)(q2.w * sc);
    a[12] = (half_t)(q3.x * sc); a[13] = (half_t)(q3.y * sc);
    a[14] = (half_t)(q3.z * sc); a[15] = (half_t)(q3.w * sc);
    afrag[kc] = a;
  }

  // D/C layout: VGPR v -> row m0 + v (+8 for upper half-wave), col = lane&15
  const int col     = lane & 15;
  const int rowbase = m0 + ((lane & 16) ? 8 : 0);

#pragma unroll
  for (int ct = 0; ct < CT_TILES; ++ct) {
    v8f c = {};
#pragma unroll
    for (int kc = 0; kc < KC_TILES; ++kc) {
      v16h b = sfrag[(ct * KC_TILES + kc) * 32 + lane];
      c = __builtin_amdgcn_wmma_f32_16x16x32_f16(
              /*neg_a=*/false, afrag[kc], /*neg_b=*/false, b,
              /*c_mod=*/(short)0, c, /*reuse_a=*/false, /*reuse_b=*/false);
    }
    const int h = ct * 16 + col;
    if (bias) {
      float bb = bias[h];
#pragma unroll
      for (int v = 0; v < 8; ++v)
        out[(size_t)(rowbase + v) * CH + h] = c[v] + bb;
    } else {
#pragma unroll
      for (int v = 0; v < 8; ++v) {
        size_t idx = (size_t)(rowbase + v) * CH + h;
        float o = out[idx] + c[v];
        if (doRelu) o = fmaxf(o, 0.0f);
        out[idx] = o;
      }
    }
  }
}

// ---------------------------------------------------------------------------
// Per-relation edge scatter: agg[dst,:] += x[src,:], cnt[dst] += 1 for edges
// of relation `rel`. 128 threads per edge (one channel each), 2 edges/block.
// ---------------------------------------------------------------------------
__global__ void __launch_bounds__(256)
scatter_rel_kernel(const int* __restrict__ src,
                   const int* __restrict__ dst,
                   const int* __restrict__ etype,
                   const float* __restrict__ x,
                   float* __restrict__ agg,
                   float* __restrict__ cntbuf,
                   int E, int rel) {
  int e = blockIdx.x * 2 + (threadIdx.x >> 7);
  if (e >= E) return;
  if (etype[e] != rel) return;
  int c = threadIdx.x & 127;
  int s = src[e], d = dst[e];
  atomicAdd(&agg[(size_t)d * CH + c], x[(size_t)s * CH + c]);
  if (c == 0) atomicAdd(&cntbuf[d], 1.0f);
}

__global__ void zero_kernel(float* __restrict__ p, long n) {
  long i = (long)blockIdx.x * blockDim.x + threadIdx.x;
  if (i < n) p[i] = 0.0f;
}

// ---------------------------------------------------------------------------
// Inputs (setup_inputs order): x[N,128] f32, edge_index[2,E] i32,
// edge_type[E] i32, W[8,128,128] f32, root[128,128] f32, bias[128] f32.
// Output: [N,128] f32.
// Workspace: B-frags (288 KB) + agg (N*128 f32, 51.2 MB) + cnt (N f32).
// ---------------------------------------------------------------------------
extern "C" void kernel_launch(void* const* d_in, const int* in_sizes, int n_in,
                              void* d_out, int out_size, void* d_ws, size_t ws_size,
                              hipStream_t stream) {
  const float* x     = (const float*)d_in[0];
  const int*   eidx  = (const int*)  d_in[1];
  const int*   etype = (const int*)  d_in[2];
  const float* W     = (const float*)d_in[3];
  const float* root  = (const float*)d_in[4];
  const float* bias  = (const float*)d_in[5];
  float* out = (float*)d_out;

  const int E = in_sizes[2];
  const int N = in_sizes[0] / CH;
  const int* src = eidx;
  const int* dst = eidx + E;

  char* ws = (char*)d_ws;
  v16h* frag = (v16h*)ws;
  size_t off = (size_t)(N_RELS + 1) * FRAGS_PER_MAT * sizeof(v16h);
  off = (off + 255) & ~(size_t)255;
  float* agg = (float*)(ws + off);
  off += (size_t)N * CH * sizeof(float);
  float* cnt = (float*)(ws + off);

  const int nRowTiles = N / 16;          // N = 100000 -> exactly 6250 tiles
  const long aggN = (long)N * CH;
  dim3 blk(256);
  const unsigned gemmGrid = (unsigned)((nRowTiles + 7) / 8);

  // 1) repack weights into WMMA B-fragment layout (f16)
  {
    int total = (N_RELS + 1) * FRAGS_PER_MAT;
    build_bfrag_kernel<<<(total + 255) / 256, blk, 0, stream>>>(W, root, frag);
  }

  // 2) root / self-loop term: out = x @ root + bias (initializes d_out)
  rgcn_wmma_gemm<<<gemmGrid, blk, 0, stream>>>(
      x, nullptr, frag + (size_t)N_RELS * FRAGS_PER_MAT, bias, out, nRowTiles, 0);

  // 3) per relation: zero, scatter (mean numerator + count), GEMM-accumulate;
  //    final relation applies the ReLU in its store path
  for (int r = 0; r < N_RELS; ++r) {
    zero_kernel<<<(unsigned)((aggN + 255) / 256), blk, 0, stream>>>(agg, aggN);
    zero_kernel<<<(unsigned)((N + 255) / 256), blk, 0, stream>>>(cnt, (long)N);
    scatter_rel_kernel<<<(unsigned)((E + 1) / 2), blk, 0, stream>>>(
        src, dst, etype, x, agg, cnt, E, r);
    rgcn_wmma_gemm<<<gemmGrid, blk, 0, stream>>>(
        agg, cnt, frag + (size_t)r * FRAGS_PER_MAT, nullptr, out, nRowTiles,
        (r == N_RELS - 1) ? 1 : 0);
  }
}
